// SpaMamba_60713657696365
// MI455X (gfx1250) — matrix operand order repaired and checked by reading, not compiled, not run on hardware
//
#include <hip/hip_runtime.h>
#include <hip/hip_bf16.h>

typedef __attribute__((ext_vector_type(16))) __bf16 v16bf;
typedef __attribute__((ext_vector_type(8)))  float  v8f;

using bf16_t = __hip_bfloat16;

__device__ __forceinline__ float  bf2f(bf16_t v) { return __bfloat162float(v); }
__device__ __forceinline__ bf16_t f2bf(float f)  { return __float2bfloat16(f); }

// ---------------------------------------------------------------------------
// constants for this problem
// ---------------------------------------------------------------------------
#define BB   4
#define CC   192
#define LL   4096          // H*W
#define ML   16384         // BB*LL
#define DI   384
#define NS   16
#define RR   12
#define NXP  48            // R + 2N = 44 padded to 48

// ---------------------------------------------------------------------------
// prep: x (B,C,H,W) f32 -> xseq (B*L, C) bf16 (row-major, K contiguous)
// ---------------------------------------------------------------------------
__global__ void k_prepx(const float* __restrict__ x, bf16_t* __restrict__ xseq) {
  int idx = blockIdx.x * blockDim.x + threadIdx.x;       // over B*C*L
  if (idx >= BB * CC * LL) return;
  int l = idx % LL;
  int c = (idx / LL) % CC;
  int b = idx / (LL * CC);
  xseq[((size_t)b * LL + l) * CC + c] = f2bf(x[idx]);
}

// ---------------------------------------------------------------------------
// prep: transpose+convert weights to bf16, K-contiguous (N x K) layout
// ---------------------------------------------------------------------------
__global__ void k_prepw(const float* __restrict__ Win,  // (192, 768)
                        const float* __restrict__ Wx,   // (384, 44)
                        const float* __restrict__ Wo,   // (384, 192)
                        bf16_t* __restrict__ WinT,      // (768, 192)
                        bf16_t* __restrict__ WxT,       // (48, 384), rows 44..47 = 0
                        bf16_t* __restrict__ WoT) {     // (192, 384)
  int i = blockIdx.x * blockDim.x + threadIdx.x;
  const int n0 = 768 * 192, n1 = NXP * 384, n2 = 192 * 384;
  if (i < n0) { int n = i / 192, k = i % 192; WinT[i] = f2bf(Win[k * 768 + n]); return; }
  i -= n0;
  if (i < n1) { int n = i / 384, k = i % 384; WxT[i] = f2bf(n < 44 ? Wx[k * 44 + n] : 0.f); return; }
  i -= n1;
  if (i < n2) { int n = i / 384, k = i % 384; WoT[i] = f2bf(Wo[k * 192 + n]); return; }
}

// ---------------------------------------------------------------------------
// bf16 WMMA GEMM: C(MxN) = A(MxK) * Bt(NxK)^T, fp32 accumulate.
// One wave computes TM x TN fragments of 16x16. Fragment loads are direct
// global loads using contiguous k order, the same for A and B:
//   A lane: row = m0+i*16+(lane&15); 16 contiguous k at kb+(lane>>4)*16
//   B lane: row = n0+j*16+(lane&15); 16 contiguous k at kb+(lane>>4)*16
// MODE 0: plain f32 store (ld = N).
// MODE 1: split epilogue (uniform per 16-col tile): n<N/2 -> xin f32,
//         n>=N/2 -> z bf16. The 384 boundary is tile-aligned so no lane
//         divergence — a single wave-uniform branch per fragment.
// ---------------------------------------------------------------------------
template <int TM, int TN, int MODE>
__global__ __launch_bounds__(128) void wmma_gemm_bf16(
    const __bf16* __restrict__ A, const __bf16* __restrict__ Bt,
    float* __restrict__ outF, bf16_t* __restrict__ outZ,
    int M, int N, int K) {
  const int lane = threadIdx.x & 31;
  const int wave = blockIdx.x * (blockDim.x >> 5) + (threadIdx.x >> 5);
  const int nTiles = N / (16 * TN);
  const int m0 = (wave / nTiles) * 16 * TM;
  const int n0 = (wave % nTiles) * 16 * TN;
  const int rlo = lane & 15;
  const int hi  = lane >> 4;

  v8f acc[TM][TN] = {};

  for (int kb = 0; kb < K; kb += 32) {
    v16bf a[TM], b[TN];
#pragma unroll
    for (int i = 0; i < TM; ++i) {
      const __bf16* ap = A + (size_t)(m0 + i * 16 + rlo) * K + kb + hi * 16;
      union { uint4 u[2]; v16bf v; } fa;
      fa.u[0] = *(const uint4*)(ap);
      fa.u[1] = *(const uint4*)(ap + 8);
      a[i] = fa.v;
    }
#pragma unroll
    for (int j = 0; j < TN; ++j) {
      const __bf16* bp = Bt + (size_t)(n0 + j * 16 + rlo) * K + kb + hi * 16;
      b[j] = *(const v16bf*)(bp);
    }
#pragma unroll
    for (int i = 0; i < TM; ++i)
#pragma unroll
      for (int j = 0; j < TN; ++j)
        acc[i][j] = __builtin_amdgcn_wmma_f32_16x16x32_bf16(
            false, a[i], false, b[j], (short)0, acc[i][j], false, false);
  }

  // C/D layout: VGPR r -> M = r + 8*(lane>=16), N = lane&15
  const int mOff = hi * 8;
#pragma unroll
  for (int i = 0; i < TM; ++i) {
#pragma unroll
    for (int j = 0; j < TN; ++j) {
      const int nTile = n0 + j * 16;           // wave-uniform
      const int mBase = m0 + i * 16 + mOff;
      if (MODE == 1 && nTile >= (N >> 1)) {
        const int half = N >> 1;
        bf16_t* zp = outZ + (size_t)mBase * half + (nTile - half + rlo);
#pragma unroll
        for (int r = 0; r < 8; ++r)
          zp[(size_t)r * half] = f2bf(acc[i][j][r]);
      } else {
        const int ld = (MODE == 1) ? (N >> 1) : N;
        float* op = outF + (size_t)mBase * ld + nTile + rlo;
#pragma unroll
        for (int r = 0; r < 8; ++r)
          op[(size_t)r * ld] = acc[i][j][r];
      }
    }
  }
}

// ---------------------------------------------------------------------------
// causal depthwise conv1d (K=4) + SiLU; emits f32 (for scan) and bf16 (GEMM A)
// ---------------------------------------------------------------------------
__global__ void k_conv(const float* __restrict__ xin, const float* __restrict__ cw,
                       const float* __restrict__ cb, float* __restrict__ xc,
                       bf16_t* __restrict__ xcbf) {
  int idx = blockIdx.x * blockDim.x + threadIdx.x;  // over ML*DI, (b,l,d)
  if (idx >= ML * DI) return;
  int d  = idx % DI;
  int bl = idx / DI;
  int l  = bl & (LL - 1);
  int b  = bl >> 12;
  float acc = cb[d];
#pragma unroll
  for (int k = 0; k < 4; ++k) {
    int ls = l + k - 3;
    if (ls >= 0) acc += xin[(size_t)(b * LL + ls) * DI + d] * cw[d * 4 + k];
  }
  float s = acc / (1.f + __expf(-acc));
  xc[idx]   = s;
  xcbf[idx] = f2bf(s);
}

// ---------------------------------------------------------------------------
// fused selective scan: thread = (b, d, n); 16 lanes share one channel d.
// dt = softplus(sum_r xdbl[:, r] * Wdt[r, d] + bdt[d])   (cross-lane reduce)
// h  = exp(dt*A) * h + dt*xc*B ;  y = sum_n h*C  (cross-lane reduce)
// out = (y + xc*D) * silu(z)  -> bf16 A-matrix of the out-projection GEMM
// ---------------------------------------------------------------------------
__global__ __launch_bounds__(256) void k_scan(
    const float* __restrict__ xdbl, const float* __restrict__ xc,
    const bf16_t* __restrict__ zbf, const float* __restrict__ Wdt,
    const float* __restrict__ bdt, const float* __restrict__ Alog,
    const float* __restrict__ Dp, bf16_t* __restrict__ yg) {
  const int b  = blockIdx.x / (DI / 16);
  const int d0 = (blockIdx.x % (DI / 16)) * 16;
  const int n  = threadIdx.x & 15;
  const int d  = d0 + (threadIdx.x >> 4);

  const float A_dn  = -__expf(Alog[d * NS + n]);
  const float wdt_n = (n < RR) ? Wdt[n * DI + d] : 0.f;
  const float bdt_d = bdt[d];
  const float D_d   = Dp[d];

  float h = 0.f;
  const size_t rowBase = (size_t)b * LL;
  for (int l = 0; l < LL; ++l) {
    const size_t row = rowBase + l;
    const float* xr = xdbl + row * NXP;
    float xv0 = xr[n];           // dt-rank slice (n>=12 multiplied by 0)
    float Bn  = xr[12 + n];
    float Cn  = xr[28 + n];
    float xcv = xc[row * DI + d];
    if (l + 1 < LL) {            // global_prefetch_b8 of next step's operands
      __builtin_prefetch(xdbl + (row + 1) * NXP, 0, 0);
      __builtin_prefetch(xc + (row + 1) * DI + d, 0, 0);
    }
    float s = xv0 * wdt_n;
    s += __shfl_xor(s, 8, 16);
    s += __shfl_xor(s, 4, 16);
    s += __shfl_xor(s, 2, 16);
    s += __shfl_xor(s, 1, 16);
    s += bdt_d;
    float dt = (s > 20.f) ? s : log1pf(__expf(s));
    float dA = __expf(dt * A_dn);
    h = dA * h + dt * xcv * Bn;
    float y = h * Cn;
    y += __shfl_xor(y, 8, 16);
    y += __shfl_xor(y, 4, 16);
    y += __shfl_xor(y, 2, 16);
    y += __shfl_xor(y, 1, 16);
    if (n == 0) {
      float zv = bf2f(zbf[row * DI + d]);
      float g  = zv / (1.f + __expf(-zv));
      yg[row * DI + d] = f2bf((y + xcv * D_d) * g);
    }
  }
}

// ---------------------------------------------------------------------------
// GroupNorm statistics: one block per (b, group); LDS tree reduction
// ---------------------------------------------------------------------------
__global__ __launch_bounds__(256) void k_gnstats(const float* __restrict__ yout,
                                                 float* __restrict__ stats) {
  const int b  = blockIdx.x >> 2;
  const int g  = blockIdx.x & 3;
  const int c0 = g * (CC / 4);
  float s = 0.f, ss = 0.f;
  for (int i = threadIdx.x; i < (CC / 4) * LL; i += 256) {
    int l = i / (CC / 4), cc = i % (CC / 4);
    float v = yout[((size_t)b * LL + l) * CC + c0 + cc];
    s += v; ss += v * v;
  }
  __shared__ float sh[256], sh2[256];
  sh[threadIdx.x] = s; sh2[threadIdx.x] = ss;
  __syncthreads();
  for (int o = 128; o > 0; o >>= 1) {
    if (threadIdx.x < o) { sh[threadIdx.x] += sh[threadIdx.x + o];
                           sh2[threadIdx.x] += sh2[threadIdx.x + o]; }
    __syncthreads();
  }
  if (threadIdx.x == 0) {
    const float inv = 1.f / ((CC / 4) * (float)LL);
    float mean = sh[0] * inv;
    float var  = sh2[0] * inv - mean * mean;
    stats[blockIdx.x * 2]     = mean;
    stats[blockIdx.x * 2 + 1] = rsqrtf(var + 1e-5f);
  }
}

// ---------------------------------------------------------------------------
// GroupNorm apply + SiLU + residual, (b,l,c) -> (b,c,h,w)
// ---------------------------------------------------------------------------
__global__ void k_finish(const float* __restrict__ yout, const float* __restrict__ x,
                         const float* __restrict__ gw, const float* __restrict__ gb,
                         const float* __restrict__ stats, float* __restrict__ out) {
  int idx = blockIdx.x * blockDim.x + threadIdx.x;   // (b,c,l) layout
  if (idx >= BB * CC * LL) return;
  int l = idx % LL;
  int c = (idx / LL) % CC;
  int b = idx / (LL * CC);
  int g = c / (CC / 4);
  float mean = stats[(b * 4 + g) * 2], rstd = stats[(b * 4 + g) * 2 + 1];
  float v  = yout[((size_t)b * LL + l) * CC + c];
  float nv = (v - mean) * rstd * gw[c] + gb[c];
  float sv = nv / (1.f + __expf(-nv));
  out[idx] = sv + x[idx];
}

// ---------------------------------------------------------------------------
extern "C" void kernel_launch(void* const* d_in, const int* in_sizes, int n_in,
                              void* d_out, int out_size, void* d_ws, size_t ws_size,
                              hipStream_t stream) {
  const float* x    = (const float*)d_in[0];
  const float* Win  = (const float*)d_in[1];
  const float* cw   = (const float*)d_in[2];
  const float* cb   = (const float*)d_in[3];
  const float* Wx   = (const float*)d_in[4];
  const float* Wdt  = (const float*)d_in[5];
  const float* bdt  = (const float*)d_in[6];
  const float* Alog = (const float*)d_in[7];
  const float* Dp   = (const float*)d_in[8];
  const float* Wo   = (const float*)d_in[9];
  const float* gw   = (const float*)d_in[10];
  const float* gb   = (const float*)d_in[11];
  float* out = (float*)d_out;

  char* ws = (char*)d_ws;
  size_t off = 0;
  auto alloc = [&](size_t bytes) -> char* {
    char* p = ws + off;
    off += (bytes + 255) & ~(size_t)255;
    return p;
  };

  bf16_t* xseq = (bf16_t*)alloc((size_t)ML * CC * 2);
  bf16_t* WinT = (bf16_t*)alloc(768 * 192 * 2);
  bf16_t* WxT  = (bf16_t*)alloc(NXP * 384 * 2);
  bf16_t* WoT  = (bf16_t*)alloc(192 * 384 * 2);
  float*  xin  = (float*) alloc((size_t)ML * DI * 4);
  bf16_t* zbf  = (bf16_t*)alloc((size_t)ML * DI * 2);
  float*  xc   = (float*) alloc((size_t)ML * DI * 4);
  bf16_t* xcbf = (bf16_t*)alloc((size_t)ML * DI * 2);
  float*  xdbl = (float*) alloc((size_t)ML * NXP * 4);
  bf16_t* yg   = (bf16_t*)alloc((size_t)ML * DI * 2);
  float*  yout = (float*) alloc((size_t)ML * CC * 4);
  float*  stats= (float*) alloc(256);

  k_prepx<<<(BB * CC * LL + 255) / 256, 256, 0, stream>>>(x, xseq);
  k_prepw<<<(768 * 192 + NXP * 384 + 192 * 384 + 255) / 256, 256, 0, stream>>>(
      Win, Wx, Wo, WinT, WxT, WoT);

  // in-proj: 16384 x 768 x 192, split epilogue -> xin f32 | z bf16
  {
    int waves = (ML / 32) * (768 / 32);      // 12288
    wmma_gemm_bf16<2, 2, 1><<<waves / 4, 128, 0, stream>>>(
        (const __bf16*)xseq, (const __bf16*)WinT, xin, zbf, ML, 768, CC);
  }

  k_conv<<<(ML * DI + 255) / 256, 256, 0, stream>>>(xin, cw, cb, xc, xcbf);

  // x-proj: 16384 x 48 x 384 (one wave covers all 48 columns)
  {
    int waves = (ML / 32) * (NXP / 48);      // 512
    wmma_gemm_bf16<2, 3, 0><<<waves / 4, 128, 0, stream>>>(
        (const __bf16*)xcbf, (const __bf16*)WxT, xdbl, nullptr, ML, NXP, DI);
  }

  k_scan<<<BB * (DI / 16), 256, 0, stream>>>(xdbl, xc, zbf, Wdt, bdt, Alog, Dp, yg);

  // out-proj: 16384 x 192 x 384
  {
    int waves = (ML / 32) * (CC / 32);       // 3072
    wmma_gemm_bf16<2, 2, 0><<<waves / 4, 128, 0, stream>>>(
        (const __bf16*)yg, (const __bf16*)WoT, yout, nullptr, ML, CC, DI);
  }

  k_gnstats<<<BB * 4, 256, 0, stream>>>(yout, stats);
  k_finish<<<(BB * CC * LL + 255) / 256, 256, 0, stream>>>(yout, x, gw, gb, stats, out);
}
